// StatePerturbationEncoder_19765439496829
// MI455X (gfx1250) — matrix operand order repaired and checked by reading, not compiled, not run on hardware
//
#include <hip/hip_runtime.h>
#include <hip/hip_bf16.h>
#include <math.h>

// ---------------------------------------------------------------------------
// Types matching the gfx1250 WMMA builtin signatures
// ---------------------------------------------------------------------------
typedef __attribute__((ext_vector_type(16))) __bf16 v16bf;
typedef __attribute__((ext_vector_type(8)))  __bf16 v8bf;
typedef __attribute__((ext_vector_type(4)))  __bf16 v4bf;
typedef __attribute__((ext_vector_type(8)))  float  v8f;
typedef __attribute__((ext_vector_type(4)))  float  v4f;

#define D_DIM   256
#define N_ROWS  32768            // B*C = 64*512
#define ROWS_PER_BLK 64
#define LDS_PAD 8                // pad row stride to 264 bf16 (528 B) -> conflict-free ds_load_b128
#define EPSF    1e-5f

__device__ __forceinline__ float gelu_exact(float x) {
    return 0.5f * x * (1.0f + erff(x * 0.70710678118654752f));
}

// ---------------------------------------------------------------------------
// prep: Wt[l][n*256+k] = bf16(W_l[k*256+n])  (transposed so B-fragments are
// contiguous 32B per lane), and zero the per-layer stat accumulators.
// ---------------------------------------------------------------------------
__global__ void prep_kernel(const float* __restrict__ W1, const float* __restrict__ W2,
                            const float* __restrict__ W3, const float* __restrict__ W4,
                            __bf16* __restrict__ Wt, float* __restrict__ stats) {
    int idx = blockIdx.x * 256 + threadIdx.x;        // 0 .. 4*65536-1
    int l = idx >> 16;
    int rem = idx & 65535;
    int k = rem >> 8;
    int n = rem & 255;
    const float* W = (l == 0) ? W1 : (l == 1) ? W2 : (l == 2) ? W3 : W4;
    Wt[(size_t)l * 65536 + (size_t)n * 256 + k] = (__bf16)W[(size_t)k * 256 + n];
    if (idx < 3 * 512) stats[idx] = 0.0f;            // sum[256],sumsq[256] x 3 layers
}

// ---------------------------------------------------------------------------
// bn_finalize: scale = gamma * rsqrt(var+eps), shift = beta - mu*scale
// ---------------------------------------------------------------------------
__global__ void bn_finalize_kernel(const float* __restrict__ stats,
                                   const float* __restrict__ gamma,
                                   const float* __restrict__ beta,
                                   float* __restrict__ bn) {
    int t = threadIdx.x;                              // 256 threads
    const float invN = 1.0f / (float)N_ROWS;
    float mu  = stats[t] * invN;
    float var = stats[256 + t] * invN - mu * mu;
    float sc  = gamma[t] * rsqrtf(var + EPSF);
    bn[t]       = sc;
    bn[256 + t] = beta[t] - mu * sc;
}

// ---------------------------------------------------------------------------
// Fused layer: [optional gather][optional BN on input] -> bf16 LDS stage ->
// WMMA GEMM (K=256) -> +bias -> exact GELU -> f32 store [-> column stats]
// Block: 256 threads = 8 waves. Block tile: 64 rows x 256 cols.
// Wave (rg,cg): rows rg*16..+15, cols cg*128..+127 (8 WMMA n-tiles).
// ---------------------------------------------------------------------------
template <bool GATHER, bool BN_IN, bool STATS>
__global__ void mlp_layer_kernel(const float* __restrict__ xin,
                                 const int*   __restrict__ ids,
                                 const float* __restrict__ table,
                                 const __bf16* __restrict__ Wt,   // [256 cols][256 k] bf16
                                 const float* __restrict__ bias,  // [256]
                                 const float* __restrict__ bn,    // scale[256], shift[256]
                                 float* __restrict__ yout,        // [N][256]
                                 float* __restrict__ stats) {     // sum[256], sumsq[256]
    __shared__ __bf16 sX[ROWS_PER_BLK][D_DIM + LDS_PAD];
    __shared__ float sSum[D_DIM];
    __shared__ float sSq[D_DIM];

    const int tid  = threadIdx.x;
    const int lane = tid & 31;
    const int wv   = tid >> 5;
    const int rg   = wv >> 1;          // 0..3  row group
    const int cg   = wv & 1;           // 0..1  col group
    const int hi   = lane >> 4;        // lane half (K-group select)
    const int lm   = lane & 15;        // M (A) or N (B/C/D) index
    const int row0 = blockIdx.x * ROWS_PER_BLK;

    if (STATS) { sSum[tid] = 0.0f; sSq[tid] = 0.0f; }

    // ---- stage 64x256 input tile into LDS as bf16 (BN applied on the fly) ----
    #pragma unroll
    for (int it = 0; it < 16; ++it) {
        int f  = it * 256 + tid;       // float4 index, 0..4095
        int r  = f >> 6;               // row in tile
        int k  = (f & 63) << 2;        // k offset
        const float* src;
        if (GATHER) src = table + (size_t)ids[row0 + r] * D_DIM + k;
        else        src = xin   + (size_t)(row0 + r) * D_DIM + k;
        v4f v = *(const v4f*)src;
        if (BN_IN) {
            v4f sc = *(const v4f*)(bn + k);
            v4f sh = *(const v4f*)(bn + 256 + k);
            v = v * sc + sh;
        }
        v4bf pv = { (__bf16)v.x, (__bf16)v.y, (__bf16)v.z, (__bf16)v.w };
        *(v4bf*)&sX[r][k] = pv;        // ds_store_b64
    }
    __syncthreads();

    // ---- WMMA main loop: 8 n-tiles, K = 256 in 8 steps of 32 ----
    v8f acc[8];
    #pragma unroll
    for (int nt = 0; nt < 8; ++nt) acc[nt] = (v8f){0,0,0,0,0,0,0,0};

    const __bf16* arow = &sX[rg * 16 + lm][0];
    #pragma unroll
    for (int kk = 0; kk < 8; ++kk) {
        const int k0 = kk * 32;
        // A fragment 16x32 bf16: lane holds K = k0 + hi*16 .. +15 (contiguous, matches B)
        v8bf a0 = *(const v8bf*)(arow + k0 + hi * 16);
        v8bf a1 = *(const v8bf*)(arow + k0 + hi * 16 + 8);
        v16bf a = __builtin_shufflevector(a0, a1, 0,1,2,3,4,5,6,7,8,9,10,11,12,13,14,15);
        #pragma unroll
        for (int nt = 0; nt < 8; ++nt) {
            const int col = cg * 128 + nt * 16 + lm;
            // B fragment 32x16 bf16: lane holds column `col`, K = k0 + hi*16 .. +15 (contiguous)
            v16bf b = *(const v16bf*)(Wt + (size_t)col * D_DIM + k0 + hi * 16);
            acc[nt] = __builtin_amdgcn_wmma_f32_16x16x32_bf16(
                false, a, false, b, (short)0, acc[nt], false, false);
        }
    }

    // ---- epilogue: bias + exact GELU + store + column stats ----
    const int rowbase = row0 + rg * 16 + hi * 8;   // C/D layout: VGPR r -> M = r + hi*8
    #pragma unroll
    for (int nt = 0; nt < 8; ++nt) {
        const int col = cg * 128 + nt * 16 + lm;
        const float bb = bias[col];
        float s = 0.0f, sq = 0.0f;
        #pragma unroll
        for (int r = 0; r < 8; ++r) {
            float g = gelu_exact(acc[nt][r] + bb);
            yout[(size_t)(rowbase + r) * D_DIM + col] = g;
            if (STATS) { s += g; sq += g * g; }
        }
        if (STATS) { atomicAdd(&sSum[col], s); atomicAdd(&sSq[col], sq); }
    }

    if (STATS) {
        __syncthreads();
        atomicAdd(&stats[tid],       sSum[tid]);
        atomicAdd(&stats[256 + tid], sSq[tid]);
    }
}

// ---------------------------------------------------------------------------
// Launch: prep -> L1(gather) -> BN1 -> L2 -> BN2 -> L3 -> BN3 -> L4(out)
// Activations ping-pong between d_ws (yA) and d_out.
// ---------------------------------------------------------------------------
extern "C" void kernel_launch(void* const* d_in, const int* in_sizes, int n_in,
                              void* d_out, int out_size, void* d_ws, size_t ws_size,
                              hipStream_t stream) {
    (void)in_sizes; (void)n_in; (void)out_size; (void)ws_size;

    const int*   ids   = (const int*)d_in[0];
    const float* table = (const float*)d_in[1];
    const float* W[4]  = { (const float*)d_in[2], (const float*)d_in[4],
                           (const float*)d_in[6], (const float*)d_in[8] };
    const float* b[4]  = { (const float*)d_in[3], (const float*)d_in[5],
                           (const float*)d_in[7], (const float*)d_in[9] };
    const float* g[3]  = { (const float*)d_in[10], (const float*)d_in[12], (const float*)d_in[14] };
    const float* be[3] = { (const float*)d_in[11], (const float*)d_in[13], (const float*)d_in[15] };

    char* ws = (char*)d_ws;
    float*  yA    = (float*)ws;                                   // 32768*256*4  = 33554432 B
    __bf16* Wt    = (__bf16*)(ws + (size_t)33554432);             // 4*65536*2    = 524288 B
    float*  stats = (float*)(ws + (size_t)33554432 + 524288);     // 3*512 floats
    float*  bn    = stats + 3 * 512;                              // 3*512 floats
    float*  out   = (float*)d_out;

    prep_kernel<<<1024, 256, 0, stream>>>(W[0], W[1], W[2], W[3], Wt, stats);

    const int nblk = N_ROWS / ROWS_PER_BLK;   // 512

    // Layer 1: gather + GEMM + GELU + stats -> yA
    mlp_layer_kernel<true, false, true><<<nblk, 256, 0, stream>>>(
        nullptr, ids, table, Wt + 0 * 65536, b[0], nullptr, yA, stats + 0 * 512);
    bn_finalize_kernel<<<1, 256, 0, stream>>>(stats + 0 * 512, g[0], be[0], bn + 0 * 512);

    // Layer 2: BN(yA) + GEMM + GELU + stats -> out
    mlp_layer_kernel<false, true, true><<<nblk, 256, 0, stream>>>(
        yA, nullptr, nullptr, Wt + 1 * 65536, b[1], bn + 0 * 512, out, stats + 1 * 512);
    bn_finalize_kernel<<<1, 256, 0, stream>>>(stats + 1 * 512, g[1], be[1], bn + 1 * 512);

    // Layer 3: BN(out) + GEMM + GELU + stats -> yA
    mlp_layer_kernel<false, true, true><<<nblk, 256, 0, stream>>>(
        out, nullptr, nullptr, Wt + 2 * 65536, b[2], bn + 1 * 512, yA, stats + 2 * 512);
    bn_finalize_kernel<<<1, 256, 0, stream>>>(stats + 2 * 512, g[2], be[2], bn + 2 * 512);

    // Layer 4: BN(yA) + GEMM + GELU -> out (no stats)
    mlp_layer_kernel<false, true, false><<<nblk, 256, 0, stream>>>(
        yA, nullptr, nullptr, Wt + 3 * 65536, b[3], bn + 2 * 512, out, nullptr);
}
